// NodeNetworkG_67937792688143
// MI455X (gfx1250) — compile-verified, matrix operand order checked by reading, not run on hardware
//
#include <hip/hip_runtime.h>
#include <hip/hip_bf16.h>

// ---------------------------------------------------------------------------
// Types for CDNA5 WMMA (wave32, 16x16x32 bf16 -> f32)
// ---------------------------------------------------------------------------
typedef __attribute__((ext_vector_type(16))) __bf16        v16bf;
typedef __attribute__((ext_vector_type(8)))  float         v8f;
typedef __attribute__((ext_vector_type(8)))  unsigned int  v8u;

#define IN_DIM   48
#define OUT_DIM  128
#define K1       144           // 3*IN_DIM
#define KT1      5             // k-tiles layer 1 (K padded 144 -> 160)
#define KT2      4             // k-tiles layer 2 (128/32)
#define NCT      8             // 128/16 column tiles
#define W1F_DW   (KT1*NCT*32*8)  // 10240 dwords
#define W2F_DW   (KT2*NCT*32*8)  // 8192 dwords

// Pin instruction order across the software pipeline (post-RA scheduler
// otherwise sinks prefetch loads back to their use, forcing loadcnt-0 waits).
#if __has_builtin(__builtin_amdgcn_sched_barrier)
#define SCHED_FENCE() __builtin_amdgcn_sched_barrier(0)
#else
#define SCHED_FENCE()
#endif

// bf16 A/B fragment K-offset for VGPR v (0..7) and lane-half h (0/1):
// k = (v>=4?16:0) + (v&3)*2 + (h?8:0)   (pairs of consecutive K in one dword)
__device__ __forceinline__ int koff(int v, int h) {
    return ((v & 4) << 2) + ((v & 3) << 1) + (h << 3);
}

__device__ __forceinline__ unsigned short f2bf(float f) {
    unsigned int u = __float_as_uint(f);
    u += 0x7FFFu + ((u >> 16) & 1u);   // round-to-nearest-even
    return (unsigned short)(u >> 16);
}

__device__ __forceinline__ unsigned int pack2bf(float lo, float hi) {
    return (unsigned int)f2bf(lo) | ((unsigned int)f2bf(hi) << 16);
}

// Native CDNA5 transcendental tanh (single V_TANH_F32). Builtin path handles
// the TRANS-op hazard; asm fallback adds the required independent op.
__device__ __forceinline__ float fast_tanh(float x) {
#if __has_builtin(__builtin_amdgcn_tanhf)
    return __builtin_amdgcn_tanhf(x);
#else
    float r;
    asm("v_tanh_f32 %0, %1\n\tv_nop" : "=v"(r) : "v"(x));
    return r;
#endif
}

__device__ __forceinline__ v8f wmma_bf16(v8u a, v8u b, v8f c) {
    return __builtin_amdgcn_wmma_f32_16x16x32_bf16(
        /*neg_a=*/false, __builtin_bit_cast(v16bf, a),
        /*neg_b=*/false, __builtin_bit_cast(v16bf, b),
        /*c_mod=*/(short)0, c, /*reuse_a=*/false, /*reuse_b=*/false);
}

// ---------------------------------------------------------------------------
// Kernel 1: zero mi/mo accumulators (contiguous region, float4 stores)
// ---------------------------------------------------------------------------
__global__ void zero_kernel(float* __restrict__ p, size_t n4) {
    size_t i = (size_t)blockIdx.x * blockDim.x + threadIdx.x;
    if (i < n4) {
        float4 z = {0.f, 0.f, 0.f, 0.f};
        *(float4*)(p + i * 4) = z;
    }
}

// ---------------------------------------------------------------------------
// Kernel 2: pre-pack W1 (K padded 144->160) and W2 into bf16 B-fragment order.
// dword index = ((kt*NCT + ct)*32 + lane)*8 + v ;  value = bf16 pair
//   B[k][n] with n = ct*16 + (lane&15), k = kt*32 + koff(v, lane>>4)
// ---------------------------------------------------------------------------
__global__ void wfrag_kernel(const float* __restrict__ W1,
                             const float* __restrict__ W2,
                             unsigned int* __restrict__ w1f,
                             unsigned int* __restrict__ w2f) {
    int idx = blockIdx.x * blockDim.x + threadIdx.x;
    if (idx < W1F_DW) {
        int v = idx & 7, lane = (idx >> 3) & 31, ct = (idx >> 8) & 7, kt = idx >> 11;
        int col = ct * 16 + (lane & 15);
        int k   = kt * 32 + koff(v, lane >> 4);
        float f0 = (k     < K1) ? W1[col * K1 + k]     : 0.f;
        float f1 = (k + 1 < K1) ? W1[col * K1 + k + 1] : 0.f;
        w1f[idx] = pack2bf(f0, f1);
    } else if (idx < W1F_DW + W2F_DW) {
        int j = idx - W1F_DW;
        int v = j & 7, lane = (j >> 3) & 31, ct = (j >> 8) & 7, kt = j >> 11;
        int col = ct * 16 + (lane & 15);
        int k   = kt * 32 + koff(v, lane >> 4);
        w2f[j] = pack2bf(W2[col * OUT_DIM + k], W2[col * OUT_DIM + k + 1]);
    }
}

// ---------------------------------------------------------------------------
// Kernel 3: edge gather + scatter-add.  12 threads per edge, float4 chunks.
//   mo[row[e]] += x[col[e]] * a[e] ;  mi[col[e]] += x[row[e]] * a[e]
// x (19MB) and mi/mo (38MB) are L2-resident (192MB L2): L2-atomic-bound phase.
// ---------------------------------------------------------------------------
__global__ void edge_scatter_kernel(const float* __restrict__ x,
                                    const int*   __restrict__ ei,
                                    const float* __restrict__ ea,
                                    float* __restrict__ mi,
                                    float* __restrict__ mo,
                                    int nEdges) {
    int tid = blockIdx.x * blockDim.x + threadIdx.x;
    int e = tid / 12;
    if (e >= nEdges) return;
    int c = (tid - e * 12) * 4;

    int r  = ei[e];            // edge_index[0][e]
    int cl = ei[nEdges + e];   // edge_index[1][e]
    float a = ea[e];

    float4 xc = *(const float4*)(x + (size_t)cl * IN_DIM + c);
    float4 xr = *(const float4*)(x + (size_t)r  * IN_DIM + c);

    float* po = mo + (size_t)r  * IN_DIM + c;
    float* pi = mi + (size_t)cl * IN_DIM + c;
    atomicAdd(po + 0, xc.x * a); atomicAdd(po + 1, xc.y * a);
    atomicAdd(po + 2, xc.z * a); atomicAdd(po + 3, xc.w * a);
    atomicAdd(pi + 0, xr.x * a); atomicAdd(pi + 1, xr.y * a);
    atomicAdd(pi + 2, xr.z * a); atomicAdd(pi + 3, xr.w * a);
}

// ---------------------------------------------------------------------------
// Kernel 4: fused 2-layer MLP with WMMA, pinned 2-deep B-frag pipeline.
// Block = 128 threads = 4 waves; each wave owns a 16-node tile.
//   A1 from [mi|mo|x] (K padded to 160), 8x5 WMMAs -> v_tanh -> bf16 h in LDS
//   A2 from LDS,                          8x4 WMMAs -> v_tanh -> f32 out
// 8 independent accumulators (no WMMA RAW chain); prefetch distance 2 with
// sched_barrier-pinned order gives "wait loadcnt<=2" before each WMMA,
// hiding one full fragment-load latency per iteration within a single wave.
// ---------------------------------------------------------------------------
__global__ void __launch_bounds__(128)
mlp_wmma_kernel(const float* __restrict__ mi, const float* __restrict__ mo,
                const float* __restrict__ x,  const float* __restrict__ b1,
                const float* __restrict__ b2,
                const unsigned int* __restrict__ w1f,
                const unsigned int* __restrict__ w2f,
                float* __restrict__ out, int nNodes) {
    __shared__ unsigned short hs[4][16 * OUT_DIM];   // per-wave bf16 h tile

    const int wave = threadIdx.x >> 5;
    const int lane = threadIdx.x & 31;
    const int half = lane >> 4;
    const int lr   = lane & 15;                      // row (A/C) or col (B/C)
    const int tileBase = (blockIdx.x * 4 + wave) * 16;
    const int node = tileBase + lr;
    const bool rowValid = node < nNodes;

    const v8u* w1v = (const v8u*)w1f;                // frag i at w1v[i*32+lane]
    const v8u* w2v = (const v8u*)w2f;

    // ---- A1 fragments straight from mi | mo | x (pairs never straddle) ----
    v8u a1[KT1];
#pragma unroll
    for (int kt = 0; kt < KT1; ++kt) {
#pragma unroll
        for (int v = 0; v < 8; ++v) {
            int k = kt * 32 + koff(v, half);         // compile-time constant
            float f0 = 0.f, f1 = 0.f;
            if (rowValid) {
                if (k < 48) {
                    const float* p = mi + (size_t)node * IN_DIM + k;
                    f0 = p[0]; f1 = p[1];
                } else if (k < 96) {
                    const float* p = mo + (size_t)node * IN_DIM + (k - 48);
                    f0 = p[0]; f1 = p[1];
                } else if (k < 144) {
                    const float* p = x + (size_t)node * IN_DIM + (k - 96);
                    f0 = p[0]; f1 = p[1];
                }
            }
            a1[kt][v] = pack2bf(f0, f1);
        }
    }

    unsigned short* myhs = hs[wave];

    // ---- Layer 1: h = tanh(M @ W1^T + b1) ----
    v8f acc1[NCT];
#pragma unroll
    for (int ct = 0; ct < NCT; ++ct) {
        float bias = b1[ct * 16 + lr];
#pragma unroll
        for (int r = 0; r < 8; ++r) acc1[ct][r] = bias;
    }
    {
        v8u bfr[2];
        bfr[0] = w1v[lane];                          // frag 0
        bfr[1] = w1v[32 + lane];                     // frag 1
        SCHED_FENCE();
#pragma unroll
        for (int i = 0; i < KT1 * NCT; ++i) {        // frag order = kt*NCT+ct
            int kt = i / NCT, ct = i % NCT;          // compile-time constants
            acc1[ct] = wmma_bf16(a1[kt], bfr[i & 1], acc1[ct]);
            if (i + 2 < KT1 * NCT)
                bfr[i & 1] = w1v[(i + 2) * 32 + lane];  // prefetch distance 2
            SCHED_FENCE();
        }
    }
#pragma unroll
    for (int ct = 0; ct < NCT; ++ct)
#pragma unroll
        for (int r = 0; r < 8; ++r) {
            int m = r + half * 8;                    // C layout: lanes16-31 -> M+8
            myhs[m * OUT_DIM + ct * 16 + lr] = f2bf(fast_tanh(acc1[ct][r]));
        }

    // order the same-wave LDS write -> read round-trip (CDNA5 split counters)
    asm volatile("s_wait_dscnt 0" ::: "memory");

    // ---- A2 fragments from LDS (packed bf16 pairs, 4B aligned) ----
    v8u a2[KT2];
#pragma unroll
    for (int kt = 0; kt < KT2; ++kt) {
#pragma unroll
        for (int v = 0; v < 8; ++v) {
            int k = kt * 32 + koff(v, half);
            a2[kt][v] = *(const unsigned int*)(myhs + lr * OUT_DIM + k);
        }
    }

    // ---- Layer 2: out = tanh(h @ W2^T + b2) ----
    v8f acc2[NCT];
#pragma unroll
    for (int ct = 0; ct < NCT; ++ct) {
        float bias = b2[ct * 16 + lr];
#pragma unroll
        for (int r = 0; r < 8; ++r) acc2[ct][r] = bias;
    }
    {
        v8u bfr[2];
        bfr[0] = w2v[lane];
        bfr[1] = w2v[32 + lane];
        SCHED_FENCE();
#pragma unroll
        for (int i = 0; i < KT2 * NCT; ++i) {
            int kt = i / NCT, ct = i % NCT;
            acc2[ct] = wmma_bf16(a2[kt], bfr[i & 1], acc2[ct]);
            if (i + 2 < KT2 * NCT)
                bfr[i & 1] = w2v[(i + 2) * 32 + lane];
            SCHED_FENCE();
        }
    }
#pragma unroll
    for (int ct = 0; ct < NCT; ++ct)
#pragma unroll
        for (int r = 0; r < 8; ++r) {
            int m = tileBase + r + half * 8;
            if (m < nNodes)
                out[(size_t)m * OUT_DIM + ct * 16 + lr] = fast_tanh(acc2[ct][r]);
        }
}

// ---------------------------------------------------------------------------
// Host launcher
// inputs: x, edge_index(int), edge_attr, W1, b1, W2, b2
// ws: mi[N*48] f32 | mo[N*48] f32 | w1frag[10240] u32 | w2frag[8192] u32
// ---------------------------------------------------------------------------
extern "C" void kernel_launch(void* const* d_in, const int* in_sizes, int n_in,
                              void* d_out, int out_size, void* d_ws, size_t ws_size,
                              hipStream_t stream) {
    const float* x  = (const float*)d_in[0];
    const int*   ei = (const int*)  d_in[1];
    const float* ea = (const float*)d_in[2];
    const float* W1 = (const float*)d_in[3];
    const float* b1 = (const float*)d_in[4];
    const float* W2 = (const float*)d_in[5];
    const float* b2 = (const float*)d_in[6];
    float* out = (float*)d_out;

    const int nNodes = in_sizes[0] / IN_DIM;
    const int nEdges = in_sizes[1] / 2;

    float* mi = (float*)d_ws;
    float* mo = mi + (size_t)nNodes * IN_DIM;
    unsigned int* w1f = (unsigned int*)(mo + (size_t)nNodes * IN_DIM);
    unsigned int* w2f = w1f + W1F_DW;

    // 1) zero accumulators (mi and mo are contiguous): N*96 floats = N*24 float4
    size_t n4 = (size_t)nNodes * (2 * IN_DIM) / 4;
    zero_kernel<<<(unsigned)((n4 + 255) / 256), 256, 0, stream>>>(mi, n4);

    // 2) pre-pack weights into bf16 WMMA B-fragment order
    wfrag_kernel<<<(W1F_DW + W2F_DW + 255) / 256, 256, 0, stream>>>(W1, W2, w1f, w2f);

    // 3) edge scatter (f32 atomics, L2-resident)
    long long etot = (long long)nEdges * 12;
    edge_scatter_kernel<<<(unsigned)((etot + 255) / 256), 256, 0, stream>>>(
        x, ei, ea, mi, mo, nEdges);

    // 4) fused WMMA MLP: 64 nodes per block (4 waves x 16-node tiles)
    int nBlocks = (nNodes + 63) / 64;
    mlp_wmma_kernel<<<nBlocks, 128, 0, stream>>>(mi, mo, x, b1, b2, w1f, w2f,
                                                 out, nNodes);
}
